// Net_72730976191040
// MI455X (gfx1250) — compile-verified, matrix operand order checked by reading, not run on hardware
//
#include <hip/hip_runtime.h>
#include <hip/hip_bf16.h>

typedef __attribute__((ext_vector_type(16))) _Float16 v16h;
typedef __attribute__((ext_vector_type(8)))  float    v8f;

#define NB   8
#define NPT  2048
#define NNODE (NB * NPT)
#define KNN  20
#define FINF 3.0e38f

// ---------------------------------------------------------------------------
// CDNA5 async global->LDS staging (ASYNCcnt) with synchronous fallback
// ---------------------------------------------------------------------------
#if defined(__has_builtin)
#if __has_builtin(__builtin_amdgcn_global_load_async_to_lds_b128) && \
    __has_builtin(__builtin_amdgcn_s_wait_asynccnt)
#define USE_ASYNC_LDS 1
#endif
#endif

typedef __attribute__((address_space(1))) void as1void;
typedef __attribute__((address_space(3))) void as3void;
typedef int int4v __attribute__((vector_size(16)));
typedef __attribute__((address_space(1))) int4v g_int4;
typedef __attribute__((address_space(3))) int4v l_int4;

__device__ inline void cp16_async(const _Float16* g, _Float16* l) {
#ifdef USE_ASYNC_LDS
    __builtin_amdgcn_global_load_async_to_lds_b128(
        (g_int4*)(as1void*)(void*)g, (l_int4*)(as3void*)(void*)l, 0, 0);
#else
    *(uint4*)l = *(const uint4*)g;
#endif
}

__device__ inline void wait_stage() {
#ifdef USE_ASYNC_LDS
    __builtin_amdgcn_s_wait_asynccnt(0);
#endif
    __syncthreads();
}

// ---------------------------------------------------------------------------
// helpers
// ---------------------------------------------------------------------------
__device__ inline void atomicMaxFloat(float* addr, float v) {
    // works with init = -inf; sign-split int trick
    if (v >= 0.0f) atomicMax((int*)addr, __float_as_int(v));
    else           atomicMin((unsigned int*)addr, __float_as_uint(v));
}

// A-fragment (16x32 f16, M x K), LDS row-major tile with stride `stride` halves.
__device__ inline v16h frag_A(const _Float16* t, int row0, int stride, int lane) {
    const _Float16* p = t + (row0 + (lane & 15)) * stride + ((lane & 16) ? 8 : 0);
    v16h f;
#pragma unroll
    for (int h = 0; h < 8; ++h) f[h] = p[h];
#pragma unroll
    for (int h = 0; h < 8; ++h) f[h + 8] = p[h + 16];
    return f;
}

// B-fragment (32x16 f16, K x N), LDS tile row-major [K][N] with stride halves.
__device__ inline v16h frag_B(const _Float16* t, int col0, int stride, int lane) {
    const _Float16* p = t + col0 + (lane & 15) + ((lane & 16) ? 8 : 0) * stride;
    v16h f;
#pragma unroll
    for (int h = 0; h < 8; ++h) f[h] = p[h * stride];
#pragma unroll
    for (int h = 0; h < 8; ++h) f[h + 8] = p[(h + 16) * stride];
    return f;
}

// ---------------------------------------------------------------------------
// prep: build f16 padded layer-1 input (pos|x|feat -> 160 cols) + sq norms
// ---------------------------------------------------------------------------
__global__ __launch_bounds__(256) void k_prep1(const float* __restrict__ pos,
                                               const float* __restrict__ xin,
                                               const float* __restrict__ feat,
                                               _Float16* __restrict__ Xh,
                                               float* __restrict__ sq) {
    int w = threadIdx.x >> 5, lane = threadIdx.x & 31;
    int node = blockIdx.x * 8 + w;
    float ss = 0.0f;
#pragma unroll
    for (int t = 0; t < 5; ++t) {
        int c = lane + 32 * t;
        float v = 0.0f;
        if (c < 3)        v = pos[node * 3 + c];
        else if (c < 6)   v = xin[node * 3 + (c - 3)];
        else if (c < 142) v = feat[node * 136 + (c - 6)];
        Xh[(size_t)node * 160 + c] = (_Float16)v;
        ss += v * v;
    }
    for (int off = 16; off; off >>= 1) ss += __shfl_xor(ss, off);
    if (lane == 0) sq[node] = ss;
}

// ---------------------------------------------------------------------------
// cast layer weight W[2C,O] f32 -> Wh[Cpad, 2O] f16
// ---------------------------------------------------------------------------
__global__ void k_cast_w_split(const float* __restrict__ Wf, _Float16* __restrict__ Wh,
                               int C, int Cpad, int O, int total) {
    int i = blockIdx.x * 256 + threadIdx.x;
    if (i >= total) return;
    int twoO = 2 * O;
    int k = i / twoO, c = i % twoO;
    float v = 0.0f;
    if (k < C) {
        int o = (c < O) ? c : (c - O);
        int r = (c < O) ? k : (C + k);
        v = Wf[r * O + o];
    }
    Wh[i] = (_Float16)v;
}

__global__ void k_cast_copy(const float* __restrict__ src, _Float16* __restrict__ dst, int n) {
    int i = blockIdx.x * 256 + threadIdx.x;
    if (i < n) dst[i] = (_Float16)src[i];
}

__global__ void k_init_neg_inf(float* __restrict__ p, int n) {
    int i = blockIdx.x * 256 + threadIdx.x;
    if (i < n) p[i] = -FINF;
}

// ---------------------------------------------------------------------------
// Distance GEMM (per batch): D[i][j] = sq[i] + sq[j] - 2 * <X_i, X_j>
// block = 256 thr (8 waves), tile 128x128, wave tile 64x32 (4x2 WMMA),
// double-buffered LDS with async global->LDS staging
// ---------------------------------------------------------------------------
__global__ __launch_bounds__(256) void k_gemm_dist(const _Float16* __restrict__ X,
                                                   int lda, int coloff,
                                                   const float* __restrict__ sq,
                                                   float* __restrict__ D,
                                                   int batchBase, int Kpad) {
    __shared__ _Float16 sI[2][128 * 40];
    __shared__ _Float16 sJ[2][128 * 40];
    int tid = threadIdx.x, lane = tid & 31, w = tid >> 5;
    int wm = w & 1, wn = w >> 1;
    int bi0 = blockIdx.x * 128, bj0 = blockIdx.y * 128;

    int r = tid >> 1, kh = (tid & 1) * 16;
    const _Float16* gI = X + (size_t)(batchBase + bi0 + r) * lda + coloff + kh;
    const _Float16* gJ = X + (size_t)(batchBase + bj0 + r) * lda + coloff + kh;
    int lofs = r * 40 + kh;

    auto stage = [&](int buf, int k0) {
        cp16_async(gI + k0, &sI[buf][lofs]);
        cp16_async(gI + k0 + 8, &sI[buf][lofs + 8]);
        cp16_async(gJ + k0, &sJ[buf][lofs]);
        cp16_async(gJ + k0 + 8, &sJ[buf][lofs + 8]);
    };

    v8f acc[4][2] = {};
    stage(0, 0);
    wait_stage();
    int buf = 0;
    for (int k0 = 0; k0 < Kpad; k0 += 32) {
        if (k0 + 32 < Kpad) stage(buf ^ 1, k0 + 32);
#pragma unroll
        for (int nt = 0; nt < 2; ++nt) {
            v16h bf = frag_A(sJ[buf], wn * 32 + nt * 16, 40, lane);  // B = X^T
#pragma unroll
            for (int mt = 0; mt < 4; ++mt) {
                v16h af = frag_A(sI[buf], wm * 64 + mt * 16, 40, lane);
                acc[mt][nt] = __builtin_amdgcn_wmma_f32_16x16x32_f16(
                    false, af, false, bf, (short)0, acc[mt][nt], false, false);
            }
        }
        wait_stage();
        buf ^= 1;
    }

    int hi = (lane & 16) ? 8 : 0;
#pragma unroll
    for (int mt = 0; mt < 4; ++mt) {
#pragma unroll
        for (int nt = 0; nt < 2; ++nt) {
            int jl = bj0 + wn * 32 + nt * 16 + (lane & 15);
            float sqj = sq[batchBase + jl];
#pragma unroll
            for (int e = 0; e < 8; ++e) {
                int il = bi0 + wm * 64 + mt * 16 + e + hi;
                D[(size_t)il * NPT + jl] = sq[batchBase + il] + sqj - 2.0f * acc[mt][nt][e];
            }
        }
    }
}

// ---------------------------------------------------------------------------
// top-20 smallest per row (one wave per row, values held in VGPRs)
// ---------------------------------------------------------------------------
__global__ __launch_bounds__(256) void k_topk20(const float* __restrict__ D,
                                                int* __restrict__ idxout) {
    int w = threadIdx.x >> 5, lane = threadIdx.x & 31;
    int row = blockIdx.x * 8 + w;  // 0..2047 (batch-local)
    const float* d = D + (size_t)row * NPT;
    float v[64];
#pragma unroll
    for (int t = 0; t < 64; ++t) v[t] = d[lane + 32 * t];

    for (int r = 0; r < KNN; ++r) {
        float best = FINF;
        int bi = 0x7fffffff;
#pragma unroll
        for (int t = 0; t < 64; ++t) {
            int c = lane + 32 * t;
            if (v[t] < best) { best = v[t]; bi = c; }
        }
        for (int off = 16; off; off >>= 1) {
            float ob = __shfl_xor(best, off);
            int   oi = __shfl_xor(bi, off);
            if (ob < best || (ob == best && oi < bi)) { best = ob; bi = oi; }
        }
        if ((bi & 31) == lane) {
            int tt = bi >> 5;
#pragma unroll
            for (int t = 0; t < 64; ++t)
                if (t == tt) v[t] = FINF;   // static-index clear, stays in VGPRs
        }
        if (lane == 0) idxout[row * KNN + r] = bi;
    }
}

// ---------------------------------------------------------------------------
// PQ GEMM: [16384 x Kpad] (f16) x [Kpad x 2O] (f16) -> PQ f32 [16384 x 2O]
// double-buffered + async staging
// ---------------------------------------------------------------------------
__global__ __launch_bounds__(256) void k_gemm_pq(const _Float16* __restrict__ A,
                                                 int lda, int coloff,
                                                 const _Float16* __restrict__ B,
                                                 int ldb, float* __restrict__ PQ,
                                                 int ldpq, int Kpad) {
    __shared__ _Float16 sA[2][128 * 40];
    __shared__ _Float16 sB[2][32 * 136];
    int tid = threadIdx.x, lane = tid & 31, w = tid >> 5;
    int wm = w & 1, wn = w >> 1;
    int rowB = blockIdx.x * 128, colB = blockIdx.y * 128;

    int ra = tid >> 1, kha = (tid & 1) * 16;
    int kb = tid >> 3, nhb = (tid & 7) * 16;
    const _Float16* gA = A + (size_t)(rowB + ra) * lda + coloff + kha;
    const _Float16* gB = B + (size_t)kb * ldb + colB + nhb;
    int la_ofs = ra * 40 + kha;
    int lb_ofs = kb * 136 + nhb;

    auto stage = [&](int buf, int k0) {
        const _Float16* pa = gA + k0;
        cp16_async(pa, &sA[buf][la_ofs]);
        cp16_async(pa + 8, &sA[buf][la_ofs + 8]);
        const _Float16* pb = gB + (size_t)k0 * ldb;
        cp16_async(pb, &sB[buf][lb_ofs]);
        cp16_async(pb + 8, &sB[buf][lb_ofs + 8]);
    };

    v8f acc[4][2] = {};
    stage(0, 0);
    wait_stage();
    int buf = 0;
    for (int k0 = 0; k0 < Kpad; k0 += 32) {
        if (k0 + 32 < Kpad) stage(buf ^ 1, k0 + 32);
#pragma unroll
        for (int nt = 0; nt < 2; ++nt) {
            v16h bf = frag_B(sB[buf], wn * 32 + nt * 16, 136, lane);
#pragma unroll
            for (int mt = 0; mt < 4; ++mt) {
                v16h af = frag_A(sA[buf], wm * 64 + mt * 16, 40, lane);
                acc[mt][nt] = __builtin_amdgcn_wmma_f32_16x16x32_f16(
                    false, af, false, bf, (short)0, acc[mt][nt], false, false);
            }
        }
        wait_stage();
        buf ^= 1;
    }

    int hi = (lane & 16) ? 8 : 0;
#pragma unroll
    for (int mt = 0; mt < 4; ++mt)
#pragma unroll
        for (int nt = 0; nt < 2; ++nt) {
            int col = colB + wn * 32 + nt * 16 + (lane & 15);
#pragma unroll
            for (int e = 0; e < 8; ++e) {
                int row = rowB + wm * 64 + mt * 16 + e + hi;
                PQ[(size_t)row * ldpq + col] = acc[mt][nt][e];
            }
        }
}

// ---------------------------------------------------------------------------
// EdgeConv epilogue: out_i = P_i + bias - Q_i + max_{j in knn(i)} Q_j
// ---------------------------------------------------------------------------
__global__ __launch_bounds__(256) void k_postlayer(const float* __restrict__ PQ,
                                                   const float* __restrict__ bias,
                                                   const int* __restrict__ idx,
                                                   int O, _Float16* __restrict__ catH,
                                                   int catoff, float* __restrict__ sq) {
    int w = threadIdx.x >> 5, lane = threadIdx.x & 31;
    int node = blockIdx.x * 8 + w;
    int base = node & ~(NPT - 1);
    const int* nb = idx + (size_t)node * KNN;
    int ld = 2 * O;
    int js[KNN];
#pragma unroll
    for (int t = 0; t < KNN; ++t) js[t] = base + nb[t];

    float ss = 0.0f;
    for (int c = lane; c < O; c += 32) {
        float m = -FINF;
#pragma unroll
        for (int t = 0; t < KNN; ++t)
            m = fmaxf(m, PQ[(size_t)js[t] * ld + O + c]);
        float val = PQ[(size_t)node * ld + c] + bias[c] - PQ[(size_t)node * ld + O + c] + m;
        catH[(size_t)node * 512 + catoff + c] = (_Float16)val;
        ss += val * val;
    }
    for (int off = 16; off; off >>= 1) ss += __shfl_xor(ss, off);
    if (lane == 0) sq[node] = ss;
}

// ---------------------------------------------------------------------------
// fc1 GEMM [16384x512]x[512x1024] + bias, fused global max-pool per batch
// ---------------------------------------------------------------------------
__global__ __launch_bounds__(256) void k_gemm_fc1_pool(const _Float16* __restrict__ A,
                                                       const _Float16* __restrict__ B,
                                                       const float* __restrict__ bias,
                                                       float* __restrict__ pooled) {
    __shared__ _Float16 sA[2][128 * 40];
    __shared__ _Float16 sB[2][32 * 136];
    __shared__ float cmax[8][128];
    int tid = threadIdx.x, lane = tid & 31, w = tid >> 5;
    int wm = w & 1, wn = w >> 1;
    int rowB = blockIdx.x * 128, colB = blockIdx.y * 128;
    int bb = rowB >> 11;  // batch index (128-row blocks never straddle a batch)

    for (int i = tid; i < 8 * 128; i += 256) ((float*)cmax)[i] = -FINF;

    int ra = tid >> 1, kha = (tid & 1) * 16;
    int kb = tid >> 3, nhb = (tid & 7) * 16;
    const _Float16* gA = A + (size_t)(rowB + ra) * 512 + kha;
    const _Float16* gB = B + (size_t)kb * 1024 + colB + nhb;
    int la_ofs = ra * 40 + kha;
    int lb_ofs = kb * 136 + nhb;

    auto stage = [&](int buf, int k0) {
        const _Float16* pa = gA + k0;
        cp16_async(pa, &sA[buf][la_ofs]);
        cp16_async(pa + 8, &sA[buf][la_ofs + 8]);
        const _Float16* pb = gB + (size_t)k0 * 1024;
        cp16_async(pb, &sB[buf][lb_ofs]);
        cp16_async(pb + 8, &sB[buf][lb_ofs + 8]);
    };

    v8f acc[4][2] = {};
    stage(0, 0);
    wait_stage();
    int buf = 0;
    for (int k0 = 0; k0 < 512; k0 += 32) {
        if (k0 + 32 < 512) stage(buf ^ 1, k0 + 32);
#pragma unroll
        for (int nt = 0; nt < 2; ++nt) {
            v16h bf = frag_B(sB[buf], wn * 32 + nt * 16, 136, lane);
#pragma unroll
            for (int mt = 0; mt < 4; ++mt) {
                v16h af = frag_A(sA[buf], wm * 64 + mt * 16, 40, lane);
                acc[mt][nt] = __builtin_amdgcn_wmma_f32_16x16x32_f16(
                    false, af, false, bf, (short)0, acc[mt][nt], false, false);
            }
        }
        wait_stage();
        buf ^= 1;
    }

    // per-lane max over this wave's 64 rows for its two columns, + bias
    int c0 = colB + wn * 32 + (lane & 15);
    float m0 = -FINF, m1 = -FINF;
#pragma unroll
    for (int mt = 0; mt < 4; ++mt)
#pragma unroll
        for (int e = 0; e < 8; ++e) {
            m0 = fmaxf(m0, acc[mt][0][e]);
            m1 = fmaxf(m1, acc[mt][1][e]);
        }
    m0 += bias[c0];
    m1 += bias[c0 + 16];
    m0 = fmaxf(m0, __shfl_xor(m0, 16));
    m1 = fmaxf(m1, __shfl_xor(m1, 16));
    __syncthreads();
    if (lane < 16) {
        cmax[w][wn * 32 + lane] = m0;
        cmax[w][wn * 32 + 16 + lane] = m1;
    }
    __syncthreads();
    if (tid < 128) {
        float m = -FINF;
#pragma unroll
        for (int ww = 0; ww < 8; ++ww) m = fmaxf(m, cmax[ww][tid]);
        atomicMaxFloat(&pooled[bb * 1024 + colB + tid], m);
    }
}

// ---------------------------------------------------------------------------
// Head: (BN-eval affine + ReLU) x2 + linear + log_softmax. One block per graph.
// ---------------------------------------------------------------------------
__global__ __launch_bounds__(256) void k_head(const float* __restrict__ pooled,
    const float* __restrict__ Wa, const float* __restrict__ ba,
    const float* __restrict__ sa, const float* __restrict__ ta,
    const float* __restrict__ Wb, const float* __restrict__ bb_,
    const float* __restrict__ sb, const float* __restrict__ tb,
    const float* __restrict__ Wc, const float* __restrict__ bc,
    float* __restrict__ out) {
    __shared__ float g[1024], h1[512], h2[256], lg[23];
    int b = blockIdx.x, tid = threadIdx.x;
    for (int i = tid; i < 1024; i += 256) g[i] = pooled[b * 1024 + i];
    __syncthreads();
    for (int o = tid; o < 512; o += 256) {
        float s = 0.0f;
        for (int k = 0; k < 1024; ++k) s += g[k] * Wa[k * 512 + o];
        s = (s + ba[o]) * sa[o] + ta[o];
        h1[o] = fmaxf(s, 0.0f);
    }
    __syncthreads();
    for (int o = tid; o < 256; o += 256) {
        float s = 0.0f;
        for (int k = 0; k < 512; ++k) s += h1[k] * Wb[k * 256 + o];
        s = (s + bb_[o]) * sb[o] + tb[o];
        h2[o] = fmaxf(s, 0.0f);
    }
    __syncthreads();
    if (tid < 23) {
        float s = 0.0f;
        for (int k = 0; k < 256; ++k) s += h2[k] * Wc[k * 23 + tid];
        lg[tid] = s + bc[tid];
    }
    __syncthreads();
    if (tid == 0) {
        float m = -FINF;
        for (int i = 0; i < 23; ++i) m = fmaxf(m, lg[i]);
        float se = 0.0f;
        for (int i = 0; i < 23; ++i) se += __expf(lg[i] - m);
        float lse = __logf(se);
        for (int i = 0; i < 23; ++i) out[b * 23 + i] = lg[i] - m - lse;
    }
}

// ---------------------------------------------------------------------------
// launch
// ---------------------------------------------------------------------------
extern "C" void kernel_launch(void* const* d_in, const int* in_sizes, int n_in,
                              void* d_out, int out_size, void* d_ws, size_t ws_size,
                              hipStream_t stream) {
    const float* pos  = (const float*)d_in[0];
    const float* xin  = (const float*)d_in[1];
    const float* feat = (const float*)d_in[2];
    const float* Wl[4] = {(const float*)d_in[4], (const float*)d_in[6],
                          (const float*)d_in[8], (const float*)d_in[10]};
    const float* bl[4] = {(const float*)d_in[5], (const float*)d_in[7],
                          (const float*)d_in[9], (const float*)d_in[11]};
    const float* Wf1  = (const float*)d_in[12];
    const float* bf1  = (const float*)d_in[13];
    float* out = (float*)d_out;

    // workspace carve-up (~75 MB)
    size_t o = 0;
    char* W = (char*)d_ws;
    auto take = [&](size_t n) { size_t r = o; o += (n + 255) & ~(size_t)255; return r; };
    _Float16* Xh1   = (_Float16*)(W + take((size_t)NNODE * 160 * 2));
    _Float16* catH  = (_Float16*)(W + take((size_t)NNODE * 512 * 2));
    float*    sq    = (float*)   (W + take((size_t)NNODE * 4));
    int*      idx   = (int*)     (W + take((size_t)NNODE * KNN * 4));
    float*    PQ    = (float*)   (W + take((size_t)NNODE * 512 * 4));
    float*    D     = (float*)   (W + take((size_t)NPT * NPT * 4));
    _Float16* Wh    = (_Float16*)(W + take((size_t)256 * 512 * 2));
    _Float16* Wf1h  = (_Float16*)(W + take((size_t)512 * 1024 * 2));
    float*    pooled= (float*)   (W + take((size_t)NB * 1024 * 4));

    // layer configs
    const int Cin[4]    = {142, 64, 64, 128};
    const int Cpad[4]   = {160, 64, 64, 128};
    const int Oc[4]     = {64, 64, 128, 256};
    const int incol[4]  = {0, 0, 64, 128};   // input column offset (in catH for L>=1)
    const int catoff[4] = {0, 64, 128, 256};

    // build padded f16 layer-1 input + sq norms
    k_prep1<<<NNODE / 8, 256, 0, stream>>>(pos, xin, feat, Xh1, sq);

    for (int L = 0; L < 4; ++L) {
        const _Float16* A = (L == 0) ? Xh1 : catH;
        int lda = (L == 0) ? 160 : 512;
        int col = incol[L];
        int Kp = Cpad[L], O = Oc[L];

        int wtotal = Kp * 2 * O;
        k_cast_w_split<<<(wtotal + 255) / 256, 256, 0, stream>>>(Wl[L], Wh, Cin[L], Kp, O, wtotal);

        for (int b = 0; b < NB; ++b) {
            k_gemm_dist<<<dim3(NPT / 128, NPT / 128), 256, 0, stream>>>(
                A, lda, col, sq, D, b * NPT, Kp);
            k_topk20<<<NPT / 8, 256, 0, stream>>>(D, idx + (size_t)b * NPT * KNN);
        }

        k_gemm_pq<<<dim3(NNODE / 128, (2 * O) / 128), 256, 0, stream>>>(
            A, lda, col, Wh, 2 * O, PQ, 2 * O, Kp);
        k_postlayer<<<NNODE / 8, 256, 0, stream>>>(PQ, bl[L], idx, O, catH, catoff[L], sq);
    }

    // fc1 + fused global max pool
    k_cast_copy<<<(512 * 1024 + 255) / 256, 256, 0, stream>>>(Wf1, Wf1h, 512 * 1024);
    k_init_neg_inf<<<(NB * 1024 + 255) / 256, 256, 0, stream>>>(pooled, NB * 1024);
    k_gemm_fc1_pool<<<dim3(NNODE / 128, 1024 / 128), 256, 0, stream>>>(catH, Wf1h, bf1, pooled);

    // head
    k_head<<<NB, 256, 0, stream>>>(pooled,
        (const float*)d_in[14], (const float*)d_in[15], (const float*)d_in[16], (const float*)d_in[17],
        (const float*)d_in[18], (const float*)d_in[19], (const float*)d_in[20], (const float*)d_in[21],
        (const float*)d_in[22], (const float*)d_in[23], out);
}
